// e3interaction_49022756716912
// MI455X (gfx1250) — matrix operand order repaired and checked by reading, not compiled, not run on hardware
//
#include <hip/hip_runtime.h>

typedef __attribute__((ext_vector_type(16))) __bf16 v16bf;
typedef __attribute__((ext_vector_type(8)))  float  v8f;

#define NBASIS      20
#define HIDDEN      64
#define WNUMEL      2304
#define RADIUS_F    5.0f
#define WAVES_PB    4            // waves per block (wave32)

// ---------------------------------------------------------------------------
// Kernel 1: per-edge radial basis + MLP1 (ReLU), store h as bf16 + unit vec.
// (emb * sqrt(NB)) @ (w1 / sqrt(NB)) -> the sqrt factors cancel exactly.
// ---------------------------------------------------------------------------
__global__ void prep_kernel(const float* __restrict__ pos,
                            const int*  __restrict__ ei,
                            const float* __restrict__ w1,
                            __bf16* __restrict__ h_bf,
                            float*  __restrict__ unit_out,
                            int E)
{
    __shared__ float s_w1[NBASIS * HIDDEN];
    for (int i = threadIdx.x; i < NBASIS * HIDDEN; i += blockDim.x) s_w1[i] = w1[i];
    __syncthreads();

    const float step  = RADIUS_F / (NBASIS + 1);
    const float istep = 1.0f / step;
    const float CONST = 1.14136f * 7.3890560989306495f;   // 1.14136 * e^2
    const float SQ2   = 1.41421356237309505f;

    for (int e = blockIdx.x * blockDim.x + threadIdx.x; e < E;
         e += gridDim.x * blockDim.x) {
        int rec = ei[e], lig = ei[E + e];
        float dx = pos[lig*3+0] - pos[rec*3+0];
        float dy = pos[lig*3+1] - pos[rec*3+1];
        float dz = pos[lig*3+2] - pos[rec*3+2];
        float r  = sqrtf(dx*dx + dy*dy + dz*dz);
        float iv = 1.0f / fmaxf(r, 1e-9f);
        unit_out[e*3+0] = dx * iv;
        unit_out[e*3+1] = dy * iv;
        unit_out[e*3+2] = dz * iv;

        float emb[NBASIS];
        #pragma unroll
        for (int i = 0; i < NBASIS; ++i) {
            float diff = (r - (float)(i + 1) * step) * istep;
            float z1 = diff + 1.0f, z2 = 1.0f - diff;
            float s1 = (z1 > 0.f) ? __expf(-1.0f / z1) : 0.f;
            float s2 = (z2 > 0.f) ? __expf(-1.0f / z2) : 0.f;
            emb[i] = CONST * s1 * s2;
        }
        for (int j = 0; j < HIDDEN; ++j) {
            float acc = 0.f;
            #pragma unroll
            for (int i = 0; i < NBASIS; ++i)
                acc = fmaf(emb[i], s_w1[i * HIDDEN + j], acc);
            h_bf[(size_t)e * HIDDEN + j] = (__bf16)(SQ2 * fmaxf(acc, 0.f));
        }
    }
}

// ---------------------------------------------------------------------------
// Kernel 2: w_mlp2 (64 x 2304 f32 row-major) -> bf16, column-major with the
// per-column 64 hid values in natural order (contiguous k = 16h + s
// A-fragment order), so each lane's A-fragment is one contiguous 32-byte load:
//   perm positions [0..15]  = hid {0..15}   (K-step 0, half 0)
//   perm positions [16..31] = hid {16..31}  (K-step 0, half 1)
//   perm positions [32..47] = hid {32..47} (K-step 1, half 0)
//   perm positions [48..63] = hid {48..63} (K-step 1, half 1)
// Folds the 1/sqrt(HIDDEN) scale.
// ---------------------------------------------------------------------------
__global__ void w2cast_kernel(const float* __restrict__ w2,
                              __bf16* __restrict__ w2a)
{
    int idx = blockIdx.x * blockDim.x + threadIdx.x;
    if (idx < HIDDEN * WNUMEL) {
        int col = idx >> 6;
        int p   = idx & 63;
        int grp = (p >> 3) & 3;                       // 8-group within 32-chunk
        int sg  = grp;                                // identity (contiguous k order)
        int hid = (p & 7) | (sg << 3) | (p & 32);
        w2a[idx] = (__bf16)(w2[(size_t)hid * WNUMEL + col] * 0.125f);
    }
}

// ---------------------------------------------------------------------------
// Kernel 3: fused GEMM (WMMA bf16) + equivariant contraction.
// A = w2 tile (M = 16 weight cols, K = 64), B = h tile (N = 16 edges).
// Lane owns D[col = n0+8*hf+r][edge = lane&15]: output component w == r is
// compile-time; contraction runs straight on the WMMA accumulators.
// ---------------------------------------------------------------------------
struct Feat {
    const float* fAs;  const float* fCs;  const float* fAv; const float* fCv;
    const float* fOP;  const float* fDot; const float* fDsc; const float* fShv;
};

template<int BLK>
__device__ __forceinline__ void contract(const v8f& d, int cb, const Feat& F,
                                         float* __restrict__ accS,
                                         float* __restrict__ accV)
{
    if constexpr (BLK == 0) {                 // a_s (x) c_s      -> out_s
        float s = F.fOP[cb >> 3];
        #pragma unroll
        for (int r = 0; r < 8; ++r) accS[r] = fmaf(d[r], s, accS[r]);
    } else if constexpr (BLK == 1) {          // inv3*<a_v,c_v>   -> out_s
        float s = F.fDot[(cb - 512) >> 3];
        #pragma unroll
        for (int r = 0; r < 8; ++r) accS[r] = fmaf(d[r], s, accS[r]);
    } else if constexpr (BLK == 2) {          // c_s              -> out_s
        float s = F.fCs[(cb - 1024) >> 3];
        #pragma unroll
        for (int r = 0; r < 8; ++r) accS[r] = fmaf(d[r], s, accS[r]);
    } else if constexpr (BLK == 3) {          // inv3*<sh_v,c_v>  -> out_s
        float s = F.fDsc[(cb - 1088) >> 3];
        #pragma unroll
        for (int r = 0; r < 8; ++r) accS[r] = fmaf(d[r], s, accS[r]);
    } else {
        float ck0, ck1, ck2;
        if constexpr (BLK == 4) {             // a_s * c_v        -> out_v
            int c2 = cb - 1152; int u = c2 >> 6, v = (c2 >> 3) & 7;
            float a = F.fAs[u];
            ck0 = a * F.fCv[v*3+0]; ck1 = a * F.fCv[v*3+1]; ck2 = a * F.fCv[v*3+2];
        } else if constexpr (BLK == 5) {      // c_v              -> out_v
            int v = (cb - 1664) >> 3;
            ck0 = F.fCv[v*3+0]; ck1 = F.fCv[v*3+1]; ck2 = F.fCv[v*3+2];
        } else if constexpr (BLK == 6) {      // a_v * c_s        -> out_v
            int c2 = cb - 1728; int u = c2 >> 6, v = (c2 >> 3) & 7;
            float cs = F.fCs[v];
            ck0 = cs * F.fAv[u*3+0]; ck1 = cs * F.fAv[u*3+1]; ck2 = cs * F.fAv[u*3+2];
        } else {                              // sh_v * c_s       -> out_v
            int v = (cb - 2240) >> 3;
            float cs = F.fCs[v];
            ck0 = cs * F.fShv[0]; ck1 = cs * F.fShv[1]; ck2 = cs * F.fShv[2];
        }
        #pragma unroll
        for (int r = 0; r < 8; ++r) {
            accV[r*3+0] = fmaf(d[r], ck0, accV[r*3+0]);
            accV[r*3+1] = fmaf(d[r], ck1, accV[r*3+1]);
            accV[r*3+2] = fmaf(d[r], ck2, accV[r*3+2]);
        }
    }
}

// Segment over an even number of N-tiles; manual 2-deep pipeline with two
// independent WMMA accumulator chains.  #pragma unroll 1 keeps the loop body
// small so register use stays bounded (no spills, no VGPR-MSB addressing).
template<int BLK>
__device__ __forceinline__ void seg(int ntB, int ntE,
    const __bf16* __restrict__ apm,    // w2a + m*64 + 16*hf
    int hf8, v16bf Bh0, v16bf Bh1, Feat F,
    float* __restrict__ accS, float* __restrict__ accV)
{
    const __bf16* ap = apm + (size_t)ntB * 1024;     // 16 cols * 64 per N-tile
    v16bf A0a = *(const v16bf*)(ap);
    v16bf A1a = *(const v16bf*)(ap + 32);
    v16bf A0b = *(const v16bf*)(ap + 1024);
    v16bf A1b = *(const v16bf*)(ap + 1024 + 32);

    #pragma unroll 1
    for (int nt = ntB; nt < ntE; nt += 2) {
        v8f d0 = {0.f,0.f,0.f,0.f,0.f,0.f,0.f,0.f};
        v8f d1 = {0.f,0.f,0.f,0.f,0.f,0.f,0.f,0.f};
        d0 = __builtin_amdgcn_wmma_f32_16x16x32_bf16(false, A0a, false, Bh0,
                                                     (short)0, d0, false, false);
        d1 = __builtin_amdgcn_wmma_f32_16x16x32_bf16(false, A0b, false, Bh0,
                                                     (short)0, d1, false, false);
        d0 = __builtin_amdgcn_wmma_f32_16x16x32_bf16(false, A1a, false, Bh1,
                                                     (short)0, d0, false, false);
        d1 = __builtin_amdgcn_wmma_f32_16x16x32_bf16(false, A1b, false, Bh1,
                                                     (short)0, d1, false, false);
        if (nt + 2 < ntE) {               // prefetch next pair of A-fragments
            const __bf16* an = apm + (size_t)(nt + 2) * 1024;
            A0a = *(const v16bf*)(an);
            A1a = *(const v16bf*)(an + 32);
            A0b = *(const v16bf*)(an + 1024);
            A1b = *(const v16bf*)(an + 1024 + 32);
        }
        contract<BLK>(d0, nt * 16 + hf8,      F, accS, accV);
        contract<BLK>(d1, nt * 16 + 16 + hf8, F, accS, accV);
    }
}

__global__ __launch_bounds__(32 * WAVES_PB)
void tp_main_kernel(const float* __restrict__ x,
                    const int*  __restrict__ ei,
                    const __bf16* __restrict__ h_bf,
                    const float* __restrict__ unit_ws,
                    const __bf16* __restrict__ w2a,
                    float* __restrict__ out,
                    int E, int n_tiles)
{
    __shared__ float sAs [WAVES_PB][16][8];
    __shared__ float sCs [WAVES_PB][16][8];
    __shared__ float sAv [WAVES_PB][16][24];
    __shared__ float sCv [WAVES_PB][16][24];
    __shared__ float sOP [WAVES_PB][16][64];   // a_s[u]*c_s[v]
    __shared__ float sDot[WAVES_PB][16][64];   // inv3 * dot(a_v[u], c_v[v])
    __shared__ float sDsc[WAVES_PB][16][8];    // inv3 * dot(sh_v, c_v[v])
    __shared__ float sShv[WAVES_PB][16][3];

    const int lane = threadIdx.x & 31;
    const int wv   = threadIdx.x >> 5;
    const int m    = lane & 15;      // edge (B column & D column)
    const int hf   = lane >> 4;      // half-wave -> which 8 weight-cols
    const float INV3  = 0.57735026918962576f;  // 1/sqrt(3)
    const float SQ3   = 1.7320508075688772f;
    const float SCALE = 1.0f / 12.0f;          // c0 == c1*inv3 == 1/12

    for (int tile = blockIdx.x * WAVES_PB + wv; tile < n_tiles;
         tile += gridDim.x * WAVES_PB) {
        const int e16 = tile * 16;
        const int e   = min(e16 + m, E - 1);

        // ---- stage per-edge features into LDS (both halves cooperate) ----
        {
            const int rec = ei[e], lig = ei[E + e];
            const float* xl = x + (size_t)lig * 32;
            const float* xr = x + (size_t)rec * 32;
            if (hf == 0) {
                #pragma unroll
                for (int u = 0; u < 8; ++u) { sAs[wv][m][u] = xl[u]; sCs[wv][m][u] = xr[u]; }
                #pragma unroll
                for (int i = 0; i < 24; ++i) { sAv[wv][m][i] = xl[8+i]; sCv[wv][m][i] = xr[8+i]; }
                float shv[3];
                #pragma unroll
                for (int k = 0; k < 3; ++k) {
                    shv[k] = SQ3 * unit_ws[(size_t)e * 3 + k];
                    sShv[wv][m][k] = shv[k];
                }
                #pragma unroll
                for (int v = 0; v < 8; ++v) {
                    float dd = 0.f;
                    #pragma unroll
                    for (int k = 0; k < 3; ++k) dd = fmaf(shv[k], xr[8 + v*3 + k], dd);
                    sDsc[wv][m][v] = INV3 * dd;
                }
            }
            // halves split the u-range of the 8x8 tables
            #pragma unroll
            for (int uu = 0; uu < 4; ++uu) {
                int u = 4 * hf + uu;
                float as = xl[u];
                #pragma unroll
                for (int v = 0; v < 8; ++v) {
                    float dd = 0.f;
                    #pragma unroll
                    for (int k = 0; k < 3; ++k)
                        dd = fmaf(xl[8 + u*3 + k], xr[8 + v*3 + k], dd);
                    sDot[wv][m][u*8 + v] = INV3 * dd;
                    sOP [wv][m][u*8 + v] = as * xr[v];
                }
            }
        }
        asm volatile("" ::: "memory");  // LDS is in-order within a wave

        // ---- B fragments: this edge's h row (loop-invariant) ----
        const __bf16* hrow = h_bf + (size_t)e * HIDDEN;
        v16bf Bh0 = *(const v16bf*)(hrow + 16 * hf);        // K =  0..31
        v16bf Bh1 = *(const v16bf*)(hrow + 32 + 16 * hf);   // K = 32..63

        float accS[8], accV[24];
        #pragma unroll
        for (int j = 0; j < 8;  ++j) accS[j] = 0.f;
        #pragma unroll
        for (int j = 0; j < 24; ++j) accV[j] = 0.f;

        Feat F;
        F.fAs  = sAs [wv][m]; F.fCs  = sCs [wv][m];
        F.fAv  = sAv [wv][m]; F.fCv  = sCv [wv][m];
        F.fOP  = sOP [wv][m]; F.fDot = sDot[wv][m];
        F.fDsc = sDsc[wv][m]; F.fShv = sShv[wv][m];

        const __bf16* apm = w2a + (size_t)m * 64 + 16 * hf;
        const int hf8 = 8 * hf;

        // 8 weight blocks -> 8 compile-time segments of the N dimension
        seg<0>(  0,  32, apm, hf8, Bh0, Bh1, F, accS, accV);
        seg<1>( 32,  64, apm, hf8, Bh0, Bh1, F, accS, accV);
        seg<2>( 64,  68, apm, hf8, Bh0, Bh1, F, accS, accV);
        seg<3>( 68,  72, apm, hf8, Bh0, Bh1, F, accS, accV);
        seg<4>( 72, 104, apm, hf8, Bh0, Bh1, F, accS, accV);
        seg<5>(104, 108, apm, hf8, Bh0, Bh1, F, accS, accV);
        seg<6>(108, 140, apm, hf8, Bh0, Bh1, F, accS, accV);
        seg<7>(140, 144, apm, hf8, Bh0, Bh1, F, accS, accV);

        // ---- combine half-waves (wave32 shuffle) and write out ----
        float res[32];
        #pragma unroll
        for (int j = 0; j < 8;  ++j) res[j]     = (accS[j] + __shfl_xor(accS[j], 16)) * SCALE;
        #pragma unroll
        for (int j = 0; j < 24; ++j) res[8 + j] = (accV[j] + __shfl_xor(accV[j], 16)) * SCALE;

        if (hf == 0 && (e16 + m) < E) {
            float* op = out + (size_t)(e16 + m) * 32;
            #pragma unroll
            for (int j = 0; j < 32; j += 4) {
                float4 t; t.x = res[j]; t.y = res[j+1]; t.z = res[j+2]; t.w = res[j+3];
                *(float4*)(op + j) = t;
            }
        }
    }
}

// ---------------------------------------------------------------------------
extern "C" void kernel_launch(void* const* d_in, const int* in_sizes, int n_in,
                              void* d_out, int out_size, void* d_ws, size_t ws_size,
                              hipStream_t stream)
{
    (void)n_in; (void)out_size; (void)ws_size;
    const float* pos = (const float*)d_in[0];
    const float* x   = (const float*)d_in[1];
    const int*   ei  = (const int*)  d_in[2];
    const float* w1  = (const float*)d_in[3];
    const float* w2  = (const float*)d_in[4];
    float* out = (float*)d_out;

    const int E = in_sizes[2] / 2;

    char* ws = (char*)d_ws;
    size_t sz_h     = (size_t)E * HIDDEN * 2;                     // bf16 h
    size_t off_unit = (sz_h + 255) & ~(size_t)255;
    size_t sz_unit  = (size_t)E * 3 * sizeof(float);
    size_t off_w2a  = (off_unit + sz_unit + 255) & ~(size_t)255;  // bf16 w2 (column-major)

    __bf16* h_bf    = (__bf16*)(ws);
    float*  unit_ws = (float*) (ws + off_unit);
    __bf16* w2a     = (__bf16*)(ws + off_w2a);

    int pgrid = (E + 255) / 256;
    prep_kernel<<<pgrid, 256, 0, stream>>>(pos, ei, w1, h_bf, unit_ws, E);

    int cgrid = (HIDDEN * WNUMEL + 255) / 256;
    w2cast_kernel<<<cgrid, 256, 0, stream>>>(w2, w2a);

    int n_tiles = (E + 15) / 16;
    int mgrid   = (n_tiles + WAVES_PB - 1) / WAVES_PB;
    tp_main_kernel<<<mgrid, 32 * WAVES_PB, 0, stream>>>(x, ei, h_bf, unit_ws,
                                                        w2a, out, E, n_tiles);
}
